// CubicPchipKANLayer_33243046871161
// MI455X (gfx1250) — matrix operand order repaired and checked by reading, not compiled, not run on hardware
//
#include <hip/hip_runtime.h>

typedef __attribute__((ext_vector_type(16))) _Float16 v16h;
typedef __attribute__((ext_vector_type(8)))  _Float16 v8h;
typedef __attribute__((ext_vector_type(8)))  float    v8f;

#define B_SZ   512
#define D_IN   256
#define D_OUT  256
#define KNOTS  64
#define TWO_K  128            // [y(64) ; m(64)]
#define RED    (D_IN * TWO_K) // 32768 reduction length
#define H_STEP (4.0f / 63.0f)
#define INV_H  (63.0f / 4.0f)

#define SEGS   8              // split-K segments
#define SEG_CH (D_IN / SEGS)  // 32 channels per segment
#define CH     2              // channels staged per LDS step
#define NSTEP  (SEG_CH / CH)  // 16 steps
#define MT     64             // M tile per block
#define NT     64             // N tile per block

// ---------------------------------------------------------------------------
// Kernel 1: per (b,i) build the 4-hot fp16 coefficient row of C (B x 32768).
// ---------------------------------------------------------------------------
__global__ __launch_bounds__(256) void coeff_kernel(const float* __restrict__ x,
                                                    _Float16* __restrict__ A) {
    int gid = blockIdx.x * 256 + threadIdx.x;   // gid = b*D_IN + i
    float xv = x[gid];
    float xc = fminf(fmaxf(xv, -2.0f), 2.0f);
    float t  = (xc + 2.0f) * INV_H;
    int   idx = (int)floorf(t);
    idx = idx < 0 ? 0 : (idx > KNOTS - 2 ? KNOTS - 2 : idx);
    float u  = t - (float)idx;
    float u2 = u * u, u3 = u2 * u;
    float h00 =  2.0f * u3 - 3.0f * u2 + 1.0f;
    float h10 =         u3 - 2.0f * u2 + u;
    float h01 = -2.0f * u3 + 3.0f * u2;
    float h11 =         u3 -        u2;

    _Float16* row = A + (size_t)gid * TWO_K;    // 256-byte aligned
    float4 z = make_float4(0.f, 0.f, 0.f, 0.f);
#pragma unroll
    for (int q = 0; q < 16; ++q) ((float4*)row)[q] = z;
    row[idx]              = (_Float16)h00;
    row[idx + 1]          = (_Float16)h01;
    row[KNOTS + idx]      = (_Float16)(h10 * H_STEP);
    row[KNOTS + idx + 1]  = (_Float16)(h11 * H_STEP);
}

// ---------------------------------------------------------------------------
// Kernel 2: per (i,o) compute PCHIP slopes and emit T row: 128 fp16 [y ; m].
// ---------------------------------------------------------------------------
__device__ __forceinline__ float pchip_inner(float d0, float d1) {
    float dn = d0 + d1;
    float hm = (__builtin_fabsf(dn) < 1e-12f) ? 0.0f : (2.0f * d0 * d1 / dn);
    return (d0 * d1 > 0.0f) ? hm : 0.0f;
}

__global__ __launch_bounds__(64) void table_kernel(const float* __restrict__ y,
                                                   _Float16* __restrict__ T) {
    int i = blockIdx.x;
    int o = blockIdx.y * 64 + threadIdx.x;
    const float* yp = y + ((size_t)i * D_OUT + o) * KNOTS;

    float yv[KNOTS];
    const float4* y4 = (const float4*)yp;
#pragma unroll
    for (int q = 0; q < KNOTS / 4; ++q) {
        float4 v = y4[q];
        yv[4*q] = v.x; yv[4*q+1] = v.y; yv[4*q+2] = v.z; yv[4*q+3] = v.w;
    }

    _Float16* dst = T + ((size_t)i * D_OUT + o) * TWO_K;

#pragma unroll
    for (int q = 0; q < 8; ++q) {
        v8h p;
#pragma unroll
        for (int j = 0; j < 8; ++j) p[j] = (_Float16)yv[q * 8 + j];
        *(v8h*)(dst + q * 8) = p;
    }

#pragma unroll
    for (int q = 0; q < 8; ++q) {
        v8h p;
#pragma unroll
        for (int j = 0; j < 8; ++j) {
            int k = q * 8 + j;
            float mv;
            if (k == 0) {
                float d0 = (yv[1] - yv[0]) * INV_H;
                float d1 = (yv[2] - yv[1]) * INV_H;
                mv = 0.5f * (3.0f * d0 - d1);
                mv = (mv * d0 <= 0.0f) ? 0.0f : mv;
                if ((d0 * d1 < 0.0f) && (__builtin_fabsf(mv) > 3.0f * __builtin_fabsf(d0)))
                    mv = 3.0f * d0;
            } else if (k == KNOTS - 1) {
                float dA = (yv[62] - yv[61]) * INV_H;
                float dB = (yv[63] - yv[62]) * INV_H;
                mv = 0.5f * (3.0f * dB - dA);
                mv = (mv * dB <= 0.0f) ? 0.0f : mv;
                if ((dB * dA < 0.0f) && (__builtin_fabsf(mv) > 3.0f * __builtin_fabsf(dB)))
                    mv = 3.0f * dB;
            } else {
                float d0 = (yv[k]     - yv[k - 1]) * INV_H;
                float d1 = (yv[k + 1] - yv[k])     * INV_H;
                mv = pchip_inner(d0, d1);
            }
            p[j] = (_Float16)mv;
        }
        *(v8h*)(dst + KNOTS + q * 8) = p;
    }
}

// ---------------------------------------------------------------------------
// CDNA5 async global -> LDS copy (16B per lane), tracked by ASYNCcnt.
// ---------------------------------------------------------------------------
__device__ __forceinline__ void async_copy16(void* lptr, const void* gptr) {
    uint32_t loff = (uint32_t)(uintptr_t)lptr;     // LDS byte offset
    uint64_t ga   = (uint64_t)(uintptr_t)gptr;
    asm volatile("global_load_async_to_lds_b128 %0, %1, off"
                 :: "v"(loff), "v"(ga) : "memory");
}
__device__ __forceinline__ void wait_async0() {
    asm volatile("s_wait_asynccnt 0x0" ::: "memory");
}

// ---------------------------------------------------------------------------
// Kernel 3: split-K WMMA GEMM. Block = 16 waves (4x4), tile 64x64, LDS
// double-buffered async staging of 2 channels (A: 64x128, B: 64x128 halves).
// Each wave: 16 steps x 8 v_wmma_f32_16x16x32_f16 on LDS fragments.
// ---------------------------------------------------------------------------
__global__ __launch_bounds__(512) void kan_wmma_kernel(const _Float16* __restrict__ A,
                                                       const _Float16* __restrict__ T,
                                                       float* __restrict__ P) {
    __shared__ __align__(32) _Float16 Atile[2][CH][MT][TWO_K];  // 64 KB
    __shared__ __align__(32) _Float16 Btile[2][CH][NT][TWO_K];  // 64 KB

    const int t    = threadIdx.x;
    const int lane = t & 31;
    const int wave = t >> 5;          // 0..15
    const int wm   = wave >> 2;       // 0..3  (M sub-tile)
    const int wn   = wave &  3;       // 0..3  (N sub-tile)
    const int mbase = blockIdx.x * MT;
    const int obase = blockIdx.y * NT;
    const int i0    = blockIdx.z * SEG_CH;

    const int aoff = (lane < 16) ? 0 : 16;   // A frag K-offset (halves)
    const int boff = (lane < 16) ? 0 : 16;   // B frag K-offset (halves)
    const int l15  = lane & 15;

    auto stage = [&](int step, int buf) {
#pragma unroll
        for (int c2 = 0; c2 < CH; ++c2) {
            int ch = i0 + step * CH + c2;
#pragma unroll
            for (int q = 0; q < 2; ++q) {
                int chunk = t + q * 512;        // 0..1023 (16B chunks)
                int r  = chunk >> 4;            // 0..63  (row / col)
                int kk = (chunk & 15) * 8;      // 0..120 (half offset)
                async_copy16(&Atile[buf][c2][r][kk],
                             A + (size_t)(mbase + r) * RED + ch * TWO_K + kk);
                async_copy16(&Btile[buf][c2][r][kk],
                             T + ((size_t)ch * D_OUT + (obase + r)) * TWO_K + kk);
            }
        }
    };

    v8f acc = {};
    stage(0, 0);
    wait_async0();
    __syncthreads();

    for (int step = 0; step < NSTEP; ++step) {
        const int buf = step & 1;
        if (step + 1 < NSTEP) stage(step + 1, buf ^ 1);
#pragma unroll
        for (int c2 = 0; c2 < CH; ++c2) {
#pragma unroll
            for (int c = 0; c < 4; ++c) {
                const _Float16* ap = &Atile[buf][c2][wm * 16 + l15][c * 32 + aoff];
                v8h alo = *(const v8h*)ap;
                v8h ahi = *(const v8h*)(ap + 8);
                v16h af;
#pragma unroll
                for (int j = 0; j < 8; ++j) { af[j] = alo[j]; af[8 + j] = ahi[j]; }
                v16h bf = *(const v16h*)&Btile[buf][c2][wn * 16 + l15][c * 32 + boff];
                acc = __builtin_amdgcn_wmma_f32_16x16x32_f16(
                    false, af, false, bf, (short)0, acc, false, false);
            }
        }
        wait_async0();      // next buffer landed
        __syncthreads();    // and everyone is done reading this buffer
    }

    // C/D layout: VGPR r -> lanes 0-15: M=r, N=lane ; lanes 16-31: M=8+r, N=lane-16
    const int mm0 = (lane < 16) ? 0 : 8;
    float* Pd = P + (size_t)blockIdx.z * B_SZ * D_OUT;
#pragma unroll
    for (int r = 0; r < 8; ++r) {
        Pd[(size_t)(mbase + wm * 16 + mm0 + r) * D_OUT + (obase + wn * 16 + l15)] = acc[r];
    }
}

// ---------------------------------------------------------------------------
// Kernel 4: deterministic reduction of split-K partials + bias.
// ---------------------------------------------------------------------------
__global__ __launch_bounds__(256) void reduce_kernel(const float* __restrict__ P,
                                                     const float* __restrict__ bias,
                                                     float* __restrict__ out) {
    int gid = blockIdx.x * 256 + threadIdx.x;   // 0..131071, (b,o) row-major
    int o = gid & (D_OUT - 1);
    float v = bias[o];
#pragma unroll
    for (int s = 0; s < SEGS; ++s) v += P[(size_t)s * B_SZ * D_OUT + gid];
    out[gid] = v;
}

// ---------------------------------------------------------------------------
extern "C" void kernel_launch(void* const* d_in, const int* in_sizes, int n_in,
                              void* d_out, int out_size, void* d_ws, size_t ws_size,
                              hipStream_t stream) {
    const float* x    = (const float*)d_in[0];   // (512, 256)
    const float* y    = (const float*)d_in[1];   // (256, 256, 64)
    const float* bias = (const float*)d_in[2];   // (256,)
    float* out        = (float*)d_out;           // (512, 256)

    char* ws = (char*)d_ws;
    _Float16* A = (_Float16*)ws;                                   // 32 MB
    _Float16* T = (_Float16*)(ws + (size_t)B_SZ * RED * 2);        // 16 MB
    float*    P = (float*)(ws + (size_t)B_SZ * RED * 2
                              + (size_t)D_IN * D_OUT * TWO_K * 2); // 8 x 0.5 MB

    coeff_kernel<<<(B_SZ * D_IN) / 256, 256, 0, stream>>>(x, A);
    table_kernel<<<dim3(D_IN, D_OUT / 64), 64, 0, stream>>>(y, T);
    kan_wmma_kernel<<<dim3(B_SZ / MT, D_OUT / NT, SEGS), 512, 0, stream>>>(A, T, P);
    reduce_kernel<<<(B_SZ * D_OUT) / 256, 256, 0, stream>>>(P, bias, out);
}
